// PR_9225589752304
// MI455X (gfx1250) — matrix operand order repaired and checked by reading, not compiled, not run on hardware
//
#include <hip/hip_runtime.h>
#include <hip/hip_bf16.h>
#include <math.h>

typedef __attribute__((ext_vector_type(16))) __bf16 v16bf;
typedef __attribute__((ext_vector_type(8)))  float  v8f;

#define N_BATCH 64
#define T_TOK   577
#define C_DIM   768
#define R_DIM   64
#define TT      32
#define NTILES  19          // ceil(577/32)
#define NTHREADS 1024
#define NWAVES  32
#define XSTS    40          // padded xst row stride (elements): 80B, 16B-aligned, low-conflict

// ---- LDS layout (bytes) ----
#define OFF_ASUM   0                         // f32[64]  (persists whole kernel)
#define OFF_CONVB  256                       // f32[64]
#define OFF_GSS    512                       // f32
#define OFF_MXS    768                       // f32[32]
#define OFF_SVALS  896                       // f32[32]
#define OFF_UNION  1024
// main phase
#define OFF_WSWZ   (OFF_UNION)               // bf16[64*768]   98304 (fragment-swizzled conv_w)
#define OFF_XS     (OFF_WSWZ + 98304)        // bf16[32*768]   49152  (t-major)
#define OFF_XST    (OFF_XS  + 49152)         // bf16[768*40]   61440  (c-major, padded)
#define OFF_LOG    (OFF_XST + 61440)         // f32[4][64][32] 32768  (qtr0 reused as a_f32)
#define OFF_ABF    (OFF_LOG + 32768)         // bf16[4*32*16]  4096   (fragment-swizzled a)
#define OFF_RED    (OFF_ABF + 4096)          // f32[32*32]     4096
// final phase (overlaps main-phase union region)
#define OFF_PLDS   (OFF_UNION)               // f32[64*768]   196608
#define OFF_RNORM  (OFF_PLDS + 196608)       // f32[64]
#define SMEM_BYTES (OFF_RED + 4096)          // 250880 bytes < 320KB

__global__ void __launch_bounds__(NTHREADS, 1)
netvlad_fused_kernel(const float* __restrict__ grids,
                     const float* __restrict__ conv_w,
                     const float* __restrict__ conv_b,
                     const float* __restrict__ centroids,
                     float* __restrict__ out) {
  extern __shared__ __align__(16) char smem[];
  float*  asum  = (float*)(smem + OFF_ASUM);
  float*  convb = (float*)(smem + OFF_CONVB);
  float*  gss   = (float*)(smem + OFF_GSS);
  float*  mxs   = (float*)(smem + OFF_MXS);
  float*  svals = (float*)(smem + OFF_SVALS);
  __bf16* wswz  = (__bf16*)(smem + OFF_WSWZ);
  __bf16* xs    = (__bf16*)(smem + OFF_XS);
  __bf16* xst   = (__bf16*)(smem + OFF_XST);
  float*  logit = (float*)(smem + OFF_LOG);     // 4 partial-sum quarters
  float*  af    = (float*)(smem + OFF_LOG);     // reuse qtr0 after logits consumed
  __bf16* abf   = (__bf16*)(smem + OFF_ABF);
  float*  red   = (float*)(smem + OFF_RED);
  float*  plds  = (float*)(smem + OFF_PLDS);
  float*  rnorm = (float*)(smem + OFF_RNORM);

  const int tid  = threadIdx.x;
  const int lane = tid & 31;
  const int wv   = tid >> 5;       // 0..31
  const int n    = blockIdx.x;
  const int lh   = lane >> 4;      // lane-half (0/1) per WMMA layouts
  const int ln16 = lane & 15;

  // ---- one-time preload ----
  if (tid < R_DIM) { convb[tid] = conv_b[tid]; asum[tid] = 0.f; }
  // conv_w -> LDS pre-swizzled into exact A-fragment order:
  // frag(mt, s) = 32 lanes x 32B contiguous chunks.
  // element (r, c): mt=r/16, s=c/32, cc=c%32 -> lh=cc>>4, e=cc&15
  for (int i = tid; i < R_DIM * C_DIM; i += NTHREADS) {
    const int r = i / C_DIM, c = i - r * C_DIM;
    const int mt = r >> 4, s = c >> 5, cc = c & 31;
    const int flh = (cc >> 4) & 1;
    const int e   = cc & 15;
    wswz[(((mt * 24 + s) * 32) + (r & 15) + 16 * flh) * 16 + e] = (__bf16)conv_w[i];
  }

  // ---- wave -> GEMM tile assignment ----
  const int g1_mt = wv & 3;          // logits r-tile
  const int g1_nt = (wv >> 2) & 1;   // logits token-tile
  const int g1_q  = wv >> 3;         // contraction quarter (0..3), 6 steps each
  const int g2_rt = wv & 3;          // p r-tile
  const int g2_cg = wv >> 2;         // 0..7, owns c-tiles cg*6 .. cg*6+5

  const v8f vzero = {0.f,0.f,0.f,0.f,0.f,0.f,0.f,0.f};
  v8f acc[6];
  #pragma unroll
  for (int i = 0; i < 6; ++i) acc[i] = vzero;

  const size_t gbase = (size_t)n * T_TOK * C_DIM;
  __syncthreads();

  for (int tile = 0; tile < NTILES; ++tile) {
    // prefetch next 96KB token tile (1024 threads x 96B) -> global_prefetch_b8
    if (tile + 1 < NTILES) {
      const float* pf = grids + gbase + (size_t)(tile + 1) * TT * C_DIM + (size_t)tid * 24;
      __builtin_prefetch(pf, 0, 1);
    }

    // ---- stage 1: load + per-token L2 normalize (1 token per wave) ----
    {
      const int tk = wv;
      const int tg = tile * TT + tk;
      const bool valid = tg < T_TOK;
      const float* g = grids + gbase + (size_t)(valid ? tg : 0) * C_DIM;
      float vals[24]; float ss = 0.f;
      #pragma unroll
      for (int i = 0; i < 24; ++i) {
        float v = valid ? g[lane + i * 32] : 0.f;
        vals[i] = v; ss += v * v;
      }
      #pragma unroll
      for (int off = 16; off > 0; off >>= 1) ss += __shfl_xor(ss, off, 32);
      const float inv = 1.f / fmaxf(sqrtf(ss), 1e-12f);
      #pragma unroll
      for (int i = 0; i < 24; ++i) {
        const int c = lane + i * 32;
        const __bf16 b = (__bf16)(vals[i] * inv);
        xs[tk * C_DIM + c] = b;                // t-major (GEMM1 B)
        xst[c * XSTS + tk] = b;                // c-major padded (GEMM2 B)
      }
    }
    __syncthreads();

    // ---- stage 2: GEMM1 logits[64,32] = conv_w * x (partial over C quarter) ----
    // batched: load 3 steps' fragments, then 3 WMMAs (one wait covers 3 steps)
    {
      v8f lacc = vzero;
      const int trow = g1_nt * 16 + ln16;
      #pragma unroll
      for (int sb = 0; sb < 2; ++sb) {
        v16bf A[3], B[3];
        #pragma unroll
        for (int s3 = 0; s3 < 3; ++s3) {
          const int s = g1_q * 6 + sb * 3 + s3;
          A[s3] = *(const v16bf*)(wswz + ((g1_mt * 24 + s) * 32 + lane) * 16);
          B[s3] = *(const v16bf*)(xs + trow * C_DIM + s * 32 + lh * 16);
        }
        #pragma unroll
        for (int s3 = 0; s3 < 3; ++s3)
          lacc = __builtin_amdgcn_wmma_f32_16x16x32_bf16(false, A[s3], false, B[s3],
                                                         (short)0, lacc, false, false);
      }
      float* dst = logit + g1_q * (R_DIM * TT);
      #pragma unroll
      for (int j = 0; j < 8; ++j) {
        const int r = g1_mt * 16 + j + lh * 8;
        const int t = g1_nt * 16 + ln16;
        dst[r * TT + t] = lacc[j];
      }
    }
    __syncthreads();

    // ---- stage 3: softmax over K=64 per token (+bias, pad masking, asum) ----
    {
      const int tok = tid & 31;
      const int grp = tid >> 5;                // 32 groups x 2 regions
      const int r0  = grp * 2;
      float l[2];
      #pragma unroll
      for (int j = 0; j < 2; ++j) {
        float v = convb[r0 + j];
        #pragma unroll
        for (int q = 0; q < 4; ++q)
          v += logit[q * (R_DIM * TT) + (r0 + j) * TT + tok];
        l[j] = v;
      }
      red[grp * 32 + tok] = fmaxf(l[0], l[1]);
      __syncthreads();
      if (tid < 32) {
        float m = red[tok];
        #pragma unroll
        for (int g2 = 1; g2 < 32; ++g2) m = fmaxf(m, red[g2 * 32 + tok]);
        mxs[tok] = m;
      }
      __syncthreads();
      const float m = mxs[tok];
      float e[2]; float s = 0.f;
      #pragma unroll
      for (int j = 0; j < 2; ++j) { e[j] = __expf(l[j] - m); s += e[j]; }
      red[grp * 32 + tok] = s;
      __syncthreads();
      if (tid < 32) {
        float t = 0.f;
        #pragma unroll
        for (int g2 = 0; g2 < 32; ++g2) t += red[g2 * 32 + tok];
        svals[tok] = 1.f / t;
      }
      __syncthreads();
      const bool valid = (tile * TT + tok) < T_TOK;
      const float invs = svals[tok];
      #pragma unroll
      for (int j = 0; j < 2; ++j) {
        const float a = valid ? e[j] * invs : 0.f;
        const int r = r0 + j;
        af[r * TT + tok] = a;                  // f32 copy (reuses logit qtr0)
        // write a directly in A-fragment-swizzled order:
        const int flh = (tok >> 4) & 1;
        const int e2  = tok & 15;
        abf[((r >> 4) * 32 + (r & 15) + 16 * flh) * 16 + e2] = (__bf16)a;
      }
      __syncthreads();
      if (tid < R_DIM) {                       // asum[r] += sum_t a[r,t]
        float s2 = 0.f;
        #pragma unroll
        for (int t = 0; t < TT; ++t) s2 += af[tid * TT + t];
        asum[tid] += s2;
      }
    }
    __syncthreads();

    // ---- stage 4: GEMM2 p[64,768] += a[64,32] * x[32,768] (register acc) ----
    // A frag: single swizzled 32B load; all 6 B frags batched before the WMMA burst
    {
      v16bf a = *(const v16bf*)(abf + (g2_rt * 32 + lane) * 16);
      v16bf B[6];
      #pragma unroll
      for (int i = 0; i < 6; ++i) {
        const int ccol = (g2_cg * 6 + i) * 16 + ln16;
        B[i] = *(const v16bf*)(xst + ccol * XSTS + lh * 16);
      }
      #pragma unroll
      for (int i = 0; i < 6; ++i)
        acc[i] = __builtin_amdgcn_wmma_f32_16x16x32_bf16(false, a, false, B[i],
                                                         (short)0, acc[i], false, false);
    }
    __syncthreads();   // xs/xst/abf reusable next tile
  }

  // ---- finalize: dump accumulators to LDS ----
  #pragma unroll
  for (int i = 0; i < 6; ++i) {
    const int ct = g2_cg * 6 + i;
    #pragma unroll
    for (int j = 0; j < 8; ++j) {
      const int r = g2_rt * 16 + j + lh * 8;
      const int c = ct * 16 + ln16;
      plds[r * C_DIM + c] = acc[i][j];
    }
  }
  if (tid == 0) *gss = 0.f;
  __syncthreads();

  // centroid subtract + intra-region L2 norm (each wave: 2 regions)
  #pragma unroll
  for (int rr = 0; rr < 2; ++rr) {
    const int r = wv * 2 + rr;
    const float av = asum[r];
    float ss = 0.f;
    #pragma unroll
    for (int i = 0; i < 24; ++i) {
      const int c = lane + i * 32;
      const float v = plds[r * C_DIM + c] - av * centroids[r * C_DIM + c];
      plds[r * C_DIM + c] = v;
      ss += v * v;
    }
    #pragma unroll
    for (int off = 16; off > 0; off >>= 1) ss += __shfl_xor(ss, off, 32);
    const float inv = 1.f / fmaxf(sqrtf(ss), 1e-12f);
    if (lane == 0) { rnorm[r] = inv; atomicAdd(gss, ss * inv * inv); }
  }
  __syncthreads();

  // global L2 norm + coalesced float4 store
  const float ginv = 1.f / fmaxf(sqrtf(*gss), 1e-12f);
  float4* out4 = (float4*)(out + (size_t)n * R_DIM * C_DIM);
  const float4* p4 = (const float4*)plds;
  for (int i = tid; i < (R_DIM * C_DIM) / 4; i += NTHREADS) {
    const int r = (i * 4) / C_DIM;             // 768 % 4 == 0: float4 within one region
    const float sc = rnorm[r] * ginv;
    float4 v = p4[i];
    v.x *= sc; v.y *= sc; v.z *= sc; v.w *= sc;
    out4[i] = v;
  }
}

extern "C" void kernel_launch(void* const* d_in, const int* in_sizes, int n_in,
                              void* d_out, int out_size, void* d_ws, size_t ws_size,
                              hipStream_t stream) {
  (void)in_sizes; (void)n_in; (void)out_size; (void)d_ws; (void)ws_size;
  const float* grids     = (const float*)d_in[0];
  const float* conv_w    = (const float*)d_in[1];
  const float* conv_b    = (const float*)d_in[2];
  const float* centroids = (const float*)d_in[3];
  float* out = (float*)d_out;

  hipFuncSetAttribute((const void*)netvlad_fused_kernel,
                      hipFuncAttributeMaxDynamicSharedMemorySize, SMEM_BYTES);
  netvlad_fused_kernel<<<N_BATCH, NTHREADS, SMEM_BYTES, stream>>>(
      grids, conv_w, conv_b, centroids, out);
}
